// Transformer_Attention_81295140979317
// MI455X (gfx1250) — matrix operand order repaired and checked by reading, not compiled, not run on hardware
//
#include <hip/hip_runtime.h>

// ---------------------------------------------------------------------------
// Transformer attention (B=4, S=4096, E=512, d=64) for MI455X / gfx1250.
// Two phases:
//   1) qkv_kernel : Q/K/V = x @ W^T via v_wmma_f32_16x16x32_bf16, bf16 out to ws
//   2) attn_kernel: flash-attention (online softmax), K/V tiles in LDS,
//                   async global->LDS copies when available, all matmuls WMMA.
// Workspace use: 3 * B*S*64 bf16 = 6 MB in d_ws.
// ---------------------------------------------------------------------------

namespace {
constexpr int kB = 4;
constexpr int kS = 4096;
constexpr int kE = 512;
constexpr int kD = 64;
}

typedef __attribute__((ext_vector_type(16))) __bf16 v16bf;
typedef __attribute__((ext_vector_type(8)))  __bf16 v8bf;
typedef __attribute__((ext_vector_type(8)))  float  v8f;
typedef __attribute__((ext_vector_type(4)))  int    v4i;

#if defined(__gfx1250__) && \
    __has_builtin(__builtin_amdgcn_global_load_async_to_lds_b128) && \
    __has_builtin(__builtin_amdgcn_s_wait_asynccnt)
#define USE_ASYNC_LDS 1
// Probe-derived prototype: (AS1 v4i* src, AS3 v4i* dst, imm offset, imm cpol)
typedef __attribute__((address_space(1))) v4i* g_v4i;
typedef __attribute__((address_space(3))) v4i* l_v4i;
#else
#define USE_ASYNC_LDS 0
#endif

__device__ __forceinline__ v8f wmma_bf16(v16bf a, v16bf b, v8f c) {
  // D = A(16x32 bf16) * B(32x16 bf16) + C(16x16 f32)
  return __builtin_amdgcn_wmma_f32_16x16x32_bf16(
      /*neg_a=*/false, a, /*neg_b=*/false, b,
      /*c_mod=*/(short)0, c, /*reuse_a=*/false, /*reuse_b=*/false);
}

__device__ __forceinline__ v16bf cat8(v8bf lo, v8bf hi) {
  return __builtin_shufflevector(lo, hi, 0, 1, 2, 3, 4, 5, 6, 7,
                                         8, 9, 10, 11, 12, 13, 14, 15);
}

// ---------------------------------------------------------------------------
// Phase 1: QKV projection.
// Grid: B*S/16 blocks of 384 threads (12 waves).
// Wave w: matrix (w/4) in {Q,K,V}, feature tile (w%4) of 16 columns.
// ---------------------------------------------------------------------------
__global__ __launch_bounds__(384)
void qkv_kernel(const float* __restrict__ x,
                const float* __restrict__ WQ,
                const float* __restrict__ WK,
                const float* __restrict__ WV,
                unsigned short* __restrict__ Qb_,
                unsigned short* __restrict__ Kb_,
                unsigned short* __restrict__ Vb_) {
  __shared__ alignas(64) __bf16 xb[16][kE + 8];  // 16 rows of x, bf16, padded

  const int tid  = threadIdx.x;
  const int tile = blockIdx.x;
  const int b    = tile / (kS / 16);
  const int row0 = (tile % (kS / 16)) * 16;

  // Cooperative stage + f32->bf16 convert of the 16x512 x tile.
  for (int i = tid; i < 16 * kE; i += 384) {
    const int r = i >> 9;
    const int c = i & (kE - 1);
    xb[r][c] = (__bf16)x[((size_t)b * kS + row0 + r) * kE + c];
  }
  __syncthreads();

  const int wave = tid >> 5;   // 0..11
  const int lane = tid & 31;
  const int mat  = wave >> 2;  // 0:Q 1:K 2:V
  const int nt   = wave & 3;   // 16-feature tile

  const float* W = (mat == 0) ? WQ : (mat == 1) ? WK : WV;
  __bf16* Out = (__bf16*)((mat == 0) ? Qb_ : (mat == 1) ? Kb_ : Vb_);

  // A-fragment addressing (16-bit A 16x32 layout):
  // lanes 0-15 : row=lane,   K = {0..7, 16..23}
  // lanes 16-31: row=lane-16,K = {8..15,24..31}
  const int arow  = lane & 15;
  const int akoff = (lane < 16) ? 0 : 8;
  // B-fragment addressing (16-bit B 32x16 layout):
  // lanes 0-15 : N=lane,    K = {0..7, 16..23}
  // lanes 16-31: N=lane-16, K = {8..15,24..31}
  const int bn    = lane & 15;
  const int bkoff = (lane < 16) ? 0 : 8;
  const float* wrow = &W[(size_t)(nt * 16 + bn) * kE + bkoff];

  v8f acc = {};
  for (int kk = 0; kk < kE; kk += 32) {
    v8bf alo = *(const v8bf*)&xb[arow][kk + akoff];
    v8bf ahi = *(const v8bf*)&xb[arow][kk + akoff + 16];
    v16bf afrag = cat8(alo, ahi);
    v16bf bfrag;
#pragma unroll
    for (int e = 0; e < 16; ++e) bfrag[e] = (__bf16)wrow[kk + e + (e >= 8 ? 8 : 0)];
    acc = wmma_bf16(afrag, bfrag, acc);
  }

  // C layout: VGPR r -> M = r (lanes 0-15) / r+8 (lanes 16-31), N = lane&15.
  const int crow = (lane < 16) ? 0 : 8;
  const int ccol = lane & 15;
#pragma unroll
  for (int r = 0; r < 8; ++r) {
    Out[((size_t)b * kS + row0 + crow + r) * kD + nt * 16 + ccol] =
        (__bf16)acc[r];
  }
}

// ---------------------------------------------------------------------------
// Phase 2: flash attention.
// Grid: B*(S/128) blocks of 256 threads (8 waves). Wave = 16 queries x d=64.
// Iterate over keys in chunks of 32: K tile + transposed V tile in LDS.
// ---------------------------------------------------------------------------
__global__ __launch_bounds__(256)
void attn_kernel(const unsigned short* __restrict__ Qb_,
                 const unsigned short* __restrict__ Kb_,
                 const unsigned short* __restrict__ Vb_,
                 float* __restrict__ out) {
  const __bf16* Qb = (const __bf16*)Qb_;
  const __bf16* Kb = (const __bf16*)Kb_;
  const __bf16* Vb = (const __bf16*)Vb_;

  __shared__ alignas(64) __bf16 Ks[32][kD + 16];  // 32 keys x 64 dims (row-major)
  __shared__ alignas(64) __bf16 Vt[kD][32 + 16];  // transposed: [v-dim][key]
  __shared__ alignas(64) __bf16 Ps[8][16][40];    // per-wave P scratch (padded)

  const int tid  = threadIdx.x;
  const int wave = tid >> 5;
  const int lane = tid & 31;
  const int b    = blockIdx.x / (kS / 128);
  const int q0   = (blockIdx.x % (kS / 128)) * 128 + wave * 16;

  // ---- Q A-fragments: 16 queries x 64 dims -> two 16x32 bf16 A-fragments.
  const int arow  = lane & 15;
  const int akoff = (lane < 16) ? 0 : 16;
  const __bf16* qrow = Qb + ((size_t)b * kS + q0 + arow) * kD;
  v16bf aq[2];
#pragma unroll
  for (int h = 0; h < 2; ++h) {
    v8bf lo = *(const v8bf*)&qrow[h * 32 + akoff];
    v8bf hi = *(const v8bf*)&qrow[h * 32 + akoff + 8];
    aq[h] = cat8(lo, hi);
  }

  v8f o[4] = {v8f{}, v8f{}, v8f{}, v8f{}};
  float m[8], l[8];
#pragma unroll
  for (int r = 0; r < 8; ++r) { m[r] = -1e30f; l[r] = 0.0f; }

  // Staging coords: 256 threads cover 32 rows x 64 cols in 8-elem chunks.
  const int srow = tid >> 3;        // 0..31 (key within tile)
  const int scol = (tid & 7) * 8;   // 0..56 (dim offset)

  const float scale = 0.125f;  // 64^-0.5

  for (int s0 = 0; s0 < kS; s0 += 32) {
    const __bf16* ksrc = Kb + ((size_t)b * kS + s0 + srow) * kD + scol;
    const __bf16* vsrc = Vb + ((size_t)b * kS + s0 + srow) * kD + scol;

#if USE_ASYNC_LDS
    __builtin_amdgcn_global_load_async_to_lds_b128(
        (g_v4i)(unsigned long long)ksrc,      // AS1 (global addr == flat addr)
        (l_v4i)(void*)&Ks[srow][scol],        // generic -> AS3 addrspacecast
        0, 0);
#else
    *(v8bf*)&Ks[srow][scol] = *(const v8bf*)ksrc;
#endif
    // Transposed V staging (scalar scatter into LDS).
    {
      v8bf vv = *(const v8bf*)vsrc;
#pragma unroll
      for (int j = 0; j < 8; ++j) Vt[scol + j][srow] = vv[j];
    }
    if (s0 + 32 < kS) {  // prefetch next tile (global_prefetch_b8)
      __builtin_prefetch(ksrc + 32 * kD, 0, 1);
      __builtin_prefetch(vsrc + 32 * kD, 0, 1);
    }
#if USE_ASYNC_LDS
    __builtin_amdgcn_s_wait_asynccnt(0);
#endif
    __syncthreads();

    // ---- scores: two 16x16 tiles (keys s0..+15 and s0+16..+31)
    const int dof = (lane < 16) ? 0 : 16;
    v8f c0 = {}, c1 = {};
    {
      const __bf16* kr0 = &Ks[lane & 15][dof];
      c0 = wmma_bf16(aq[0], *(const v16bf*)&kr0[0], c0);
      c0 = wmma_bf16(aq[1], *(const v16bf*)&kr0[32], c0);
      const __bf16* kr1 = &Ks[16 + (lane & 15)][dof];
      c1 = wmma_bf16(aq[0], *(const v16bf*)&kr1[0], c1);
      c1 = wmma_bf16(aq[1], *(const v16bf*)&kr1[32], c1);
    }

    // ---- online softmax (row = r + 8*(lane>=16); cols across 16-lane half)
#pragma unroll
    for (int r = 0; r < 8; ++r) {
      float s0v = c0[r] * scale;
      float s1v = c1[r] * scale;
      float t = fmaxf(s0v, s1v);
#pragma unroll
      for (int mk = 1; mk < 16; mk <<= 1) t = fmaxf(t, __shfl_xor(t, mk, 32));
      const float mn    = fmaxf(m[r], t);
      const float alpha = __expf(m[r] - mn);
      m[r] = mn;
      const float p0 = __expf(s0v - mn);
      const float p1 = __expf(s1v - mn);
      float rs = p0 + p1;
#pragma unroll
      for (int mk = 1; mk < 16; mk <<= 1) rs += __shfl_xor(rs, mk, 32);
      l[r] = l[r] * alpha + rs;
      o[0][r] *= alpha; o[1][r] *= alpha; o[2][r] *= alpha; o[3][r] *= alpha;
      // C-layout -> LDS (row-major 16x32 P tile, bf16)
      const int prow = r + ((lane < 16) ? 0 : 8);
      const int pcol = lane & 15;
      Ps[wave][prow][pcol]      = (__bf16)p0;
      Ps[wave][prow][16 + pcol] = (__bf16)p1;
    }

    // ---- P A-fragment (16x32) from per-wave LDS scratch
    const int pkoff = (lane < 16) ? 0 : 16;
    v8bf plo = *(const v8bf*)&Ps[wave][arow][pkoff];
    v8bf phi = *(const v8bf*)&Ps[wave][arow][pkoff + 8];
    v16bf pa = cat8(plo, phi);

    // ---- PV: 4 output tiles of 16 v-dims each; B from transposed V tile.
    const int vn    = lane & 15;
    const int vkoff = (lane < 16) ? 0 : 16;
#pragma unroll
    for (int t4 = 0; t4 < 4; ++t4) {
      v16bf vb = *(const v16bf*)&Vt[t4 * 16 + vn][vkoff];
      o[t4] = wmma_bf16(pa, vb, o[t4]);
    }

    __syncthreads();  // protect Ks/Vt before next staging round
  }

  // ---- epilogue: out = O / l  (f32)
  const int orow = (lane < 16) ? 0 : 8;
  const int ocol = lane & 15;
#pragma unroll
  for (int r = 0; r < 8; ++r) {
    const float inv = 1.0f / l[r];
    const size_t base = ((size_t)b * kS + q0 + orow + r) * kD;
    out[base + 0  + ocol] = o[0][r] * inv;
    out[base + 16 + ocol] = o[1][r] * inv;
    out[base + 32 + ocol] = o[2][r] * inv;
    out[base + 48 + ocol] = o[3][r] * inv;
  }
}

// ---------------------------------------------------------------------------
extern "C" void kernel_launch(void* const* d_in, const int* in_sizes, int n_in,
                              void* d_out, int out_size, void* d_ws,
                              size_t ws_size, hipStream_t stream) {
  (void)in_sizes; (void)n_in; (void)out_size; (void)ws_size;
  const float* x  = (const float*)d_in[0];
  const float* WQ = (const float*)d_in[1];
  const float* WK = (const float*)d_in[2];
  const float* WV = (const float*)d_in[3];
  float* out = (float*)d_out;

  // bf16 Q/K/V in workspace: 3 * 4*4096*64 * 2B = 6 MB
  unsigned short* Qb = (unsigned short*)d_ws;
  unsigned short* Kb = Qb + (size_t)kB * kS * kD;
  unsigned short* Vb = Kb + (size_t)kB * kS * kD;

  qkv_kernel<<<kB * kS / 16, 384, 0, stream>>>(x, WQ, WK, WV, Qb, Kb, Vb);
  attn_kernel<<<kB * (kS / 128), 256, 0, stream>>>(Qb, Kb, Vb, out);
}
